// DualDGCNN7_37357625540997
// MI455X (gfx1250) — matrix operand order repaired and checked by reading, not compiled, not run on hardware
//
#include <hip/hip_runtime.h>
#include <hip/hip_bf16.h>
#include <cstdint>

#define BB 8
#define NP 1024
#define KNN 20
#define FEATS_CH 760
#define EMB 1024
#define NCLS 40

typedef __attribute__((ext_vector_type(16))) __bf16 bf16x16;
typedef __attribute__((ext_vector_type(8)))  float  f32x8;

__device__ __forceinline__ unsigned short f2bf(float f) {
  union { float f; unsigned u; } v; v.f = f;
  unsigned r = (v.u + 0x7FFFu + ((v.u >> 16) & 1u)) >> 16;
  return (unsigned short)r;
}
__device__ __forceinline__ float leakyf(float x) { return x >= 0.f ? x : 0.2f * x; }
__device__ __forceinline__ void atomicMaxF(float* a, float v) {
  unsigned* ua = (unsigned*)a;
  unsigned old = *ua;
  while (__uint_as_float(old) < v) {
    unsigned assumed = old;
    old = atomicCAS(ua, assumed, __float_as_uint(v));
    if (old == assumed) break;
  }
}
// Pack two 16-byte chunks (8 dwords = 16 bf16) into a WMMA fragment register set.
__device__ __forceinline__ bf16x16 frag_cast(uint4 a, uint4 b) {
  union { unsigned u[8]; bf16x16 v; } x;
  x.u[0] = a.x; x.u[1] = a.y; x.u[2] = a.z; x.u[3] = a.w;
  x.u[4] = b.x; x.u[5] = b.y; x.u[6] = b.z; x.u[7] = b.w;
  return x.v;
}
// CDNA5 async global->LDS DMA (ASYNCcnt-tracked), 16B per lane.
// Flat-address truncation rule (ISA 10.2): low 32 bits of a generic LDS
// pointer are the LDS address, so pass them in the VDST operand.
__device__ __forceinline__ void async_copy_b128(unsigned lds_addr, const void* gptr) {
  asm volatile("global_load_async_to_lds_b128 %0, %1, off"
               :: "v"(lds_addr), "v"((unsigned long long)(uintptr_t)gptr)
               : "memory");
}
__device__ __forceinline__ void wait_async() {
  asm volatile("s_wait_asynccnt 0" ::: "memory");
}

#define GT_M 128
#define GT_N 64
#define GT_K 32
#define LDS_STRIDE 48  // halves; 96B row stride (16B multiple, conflict-skewed)

// ---------------------------------------------------------------------------
// Generic bf16 GEMM: C(M x Nn) = A(M x Kd) * B(Kd x Nn), row-major, fp32 out.
// M % 128 == 0, Nn % 64 == 0, Kd % 32 == 0 (buffers zero-padded + 256B aligned).
// A tile staged with async LDS DMA; B tile stored transposed (n-major) so both
// fragment types load as two ds_load_b128 with zero repacking.
// ---------------------------------------------------------------------------
__global__ __launch_bounds__(256) void gemm_bf16_kernel(
    const unsigned short* __restrict__ A, const unsigned short* __restrict__ Bm,
    float* __restrict__ C, int M, int Nn, int Kd,
    long long sA, long long sB, long long sC)
{
  __shared__ alignas(16) unsigned short As[GT_M][LDS_STRIDE];
  __shared__ alignas(16) unsigned short Bt[GT_N][LDS_STRIDE];
  const int tid = threadIdx.x;
  const int lane = tid & 31;
  const int wave = tid >> 5;
  const int wm = (wave >> 1) * 32;
  const int wn = (wave & 1) * 32;
  const long long bA = (long long)blockIdx.z * sA;
  const long long bB = (long long)blockIdx.z * sB;
  const long long bC = (long long)blockIdx.z * sC;
  const int mT = blockIdx.y * GT_M;
  const int nT = blockIdx.x * GT_N;
  (void)M;

  const f32x8 vzero = {0.f,0.f,0.f,0.f,0.f,0.f,0.f,0.f};
  f32x8 acc[2][2];
#pragma unroll
  for (int i = 0; i < 2; ++i)
#pragma unroll
    for (int j = 0; j < 2; ++j) acc[i][j] = vzero;

  for (int k0 = 0; k0 < Kd; k0 += GT_K) {
    // A tile: 128 rows x 32 halves = 512 x 16B async DMA transfers
    for (int i = tid; i < GT_M * 4; i += 256) {
      int m = i >> 2, seg = i & 3;
      async_copy_b128((unsigned)(uintptr_t)&As[m][seg * 8],
                      A + bA + (long long)(mT + m) * Kd + k0 + seg * 8);
    }
    // B tile: 32 k x 64 n -> transposed Bt[n][k]; 2 cols per thread
    for (int i = tid; i < GT_K * (GT_N / 2); i += 256) {
      int k = i >> 5, n2 = (i & 31) * 2;
      unsigned w = *(const unsigned*)(Bm + bB + (long long)(k0 + k) * Nn + nT + n2);
      Bt[n2][k]     = (unsigned short)(w & 0xffffu);
      Bt[n2 + 1][k] = (unsigned short)(w >> 16);
    }
    wait_async();
    __syncthreads();
    if (k0 + GT_K < Kd)  // CDNA5 global_prefetch_b8 hint for next A tile
      __builtin_prefetch(&A[bA + (long long)(mT + (tid >> 1)) * Kd + k0 + GT_K], 0, 3);

    bf16x16 af[2], bfr[2];
#pragma unroll
    for (int i = 0; i < 2; ++i) {
      // A frag: lanes 0-15 M rows, K {bk..bk+7, 16+bk..16+bk+7}, bk = 8*(lane>>4)
      int m = wm + i * 16 + (lane & 15);
      int bk = (lane >> 4) * 8;
      af[i] = frag_cast(*(const uint4*)&As[m][bk], *(const uint4*)&As[m][16 + bk]);
      // B frag: lane%16 = N col, K {kb..kb+7, 16+kb..16+kb+7}, kb = 8*(lane>>4)
      int n = wn + i * 16 + (lane & 15);
      int kb = (lane >> 4) * 8;
      bfr[i] = frag_cast(*(const uint4*)&Bt[n][kb], *(const uint4*)&Bt[n][16 + kb]);
    }
#pragma unroll
    for (int i = 0; i < 2; ++i)
#pragma unroll
      for (int j = 0; j < 2; ++j)
        acc[i][j] = __builtin_amdgcn_wmma_f32_16x16x32_bf16(
            false, af[i], false, bfr[j], (short)0, acc[i][j], false, false);
    __syncthreads();
  }

#pragma unroll
  for (int i = 0; i < 2; ++i)
#pragma unroll
    for (int j = 0; j < 2; ++j) {
      int col = nT + wn + j * 16 + (lane & 15);
#pragma unroll
      for (int t = 0; t < 8; ++t) {
        int row = mT + wm + i * 16 + t + 8 * (lane >> 4);
        C[bC + (long long)row * Nn + col] = acc[i][j][t];
      }
    }
}

// ---------------------------------------------------------------------------
// Edge-conv: y[o, (n,k)] = W(O x 2C) . f(2C x N*K), edge features gathered on
// the fly into a transposed LDS B-tile. pass1: channel sum/sumsq stats.
// pass2: BN + leaky + atomic max over k -> yout(B,O,N).
// ---------------------------------------------------------------------------
__global__ __launch_bounds__(256) void edgeconv_kernel(
    const float* __restrict__ h, const int* __restrict__ idx,
    const unsigned short* __restrict__ Wbf,  // 128 x 256 zero-padded
    int C, int O, int KdP, int pass,
    float* __restrict__ gsum, float* __restrict__ gsumsq,
    const float* __restrict__ bn_mean, const float* __restrict__ bn_inv,
    const float* __restrict__ bn_g, const float* __restrict__ bn_b,
    float* __restrict__ yout)
{
  __shared__ alignas(16) unsigned short As[GT_M][LDS_STRIDE];
  __shared__ alignas(16) unsigned short Bt[GT_N][LDS_STRIDE];
  const int tid = threadIdx.x, lane = tid & 31, wave = tid >> 5;
  const int b = blockIdx.y;
  const int colT = blockIdx.x * 64;
  const int m0 = wave * 16;
  const int Mp = (O + 15) & ~15;
  const int C2 = 2 * C;
  const float* hb = h + (long long)b * C * NP;
  const int* ib = idx + (long long)b * NP * KNN;

  const f32x8 vzero = {0.f,0.f,0.f,0.f,0.f,0.f,0.f,0.f};
  f32x8 acc[4];
#pragma unroll
  for (int j = 0; j < 4; ++j) acc[j] = vzero;

  for (int k0 = 0; k0 < KdP; k0 += GT_K) {
    // Weight tile: async LDS DMA (Wbf rows are 512B, fully aligned)
    for (int i = tid; i < GT_M * 4; i += 256) {
      int m = i >> 2, seg = i & 3;
      async_copy_b128((unsigned)(uintptr_t)&As[m][seg * 8],
                      Wbf + m * 256 + k0 + seg * 8);
    }
    // Edge-feature tile, gathered + converted, stored transposed Bt[col][k]
    for (int i = tid; i < GT_K * 64; i += 256) {
      int kk = i >> 6, cc = i & 63;
      int cp = k0 + kk;
      int col = colT + cc;
      int n = col / KNN, kn = col % KNN;
      float v = 0.f;
      if (cp < C) {
        int nb = ib[n * KNN + kn];
        v = hb[cp * NP + nb] - hb[cp * NP + n];
      } else if (cp < C2) {
        v = hb[(cp - C) * NP + n];
      }
      Bt[cc][kk] = f2bf(v);
    }
    wait_async();
    __syncthreads();
    if (m0 < Mp) {  // wave-uniform branch; EXEC stays all-ones for WMMA
      int m = m0 + (lane & 15);
      int bk = (lane >> 4) * 8;
      bf16x16 af = frag_cast(*(const uint4*)&As[m][bk], *(const uint4*)&As[m][16 + bk]);
#pragma unroll
      for (int j = 0; j < 4; ++j) {
        int n = j * 16 + (lane & 15);
        int kb = (lane >> 4) * 8;
        bf16x16 bfr = frag_cast(*(const uint4*)&Bt[n][kb], *(const uint4*)&Bt[n][16 + kb]);
        acc[j] = __builtin_amdgcn_wmma_f32_16x16x32_bf16(
            false, af, false, bfr, (short)0, acc[j], false, false);
      }
    }
    __syncthreads();
  }
  if (m0 >= Mp) return;

  if (pass == 1) {
#pragma unroll
    for (int t = 0; t < 8; ++t) {
      int o = m0 + t + 8 * (lane >> 4);
      float s = 0.f, sq = 0.f;
#pragma unroll
      for (int j = 0; j < 4; ++j) { float v = acc[j][t]; s += v; sq += v * v; }
#pragma unroll
      for (int off = 1; off < 16; off <<= 1) {
        s += __shfl_xor(s, off);
        sq += __shfl_xor(sq, off);
      }
      if ((lane & 15) == 0 && o < O) {
        atomicAdd(&gsum[o], s);
        atomicAdd(&gsumsq[o], sq);
      }
    }
  } else {
#pragma unroll
    for (int j = 0; j < 4; ++j) {
      int col = colT + j * 16 + (lane & 15);
      int n = col / KNN;
#pragma unroll
      for (int t = 0; t < 8; ++t) {
        int o = m0 + t + 8 * (lane >> 4);
        if (o < O) {
          float v = bn_g[o] * (acc[j][t] - bn_mean[o]) * bn_inv[o] + bn_b[o];
          v = leakyf(v);
          atomicMaxF(&yout[((long long)b * O + o) * NP + n], v);
        }
      }
    }
  }
}

// ----------------------------- small kernels -------------------------------
__global__ void convgram_kernel(const float* __restrict__ h,
                                unsigned short* __restrict__ hbf,
                                unsigned short* __restrict__ hTbf, int C, int Cp)
{
  long long i = (long long)blockIdx.x * 256 + threadIdx.x;
  long long tot = (long long)BB * Cp * NP;
  if (i >= tot) return;
  int n = (int)(i % NP);
  long long r = i / NP;
  int c = (int)(r % Cp);
  int b = (int)(r / Cp);
  float v = (c < C) ? h[((long long)b * C + c) * NP + n] : 0.f;
  unsigned short u = f2bf(v);
  hbf[((long long)b * Cp + c) * NP + n] = u;
  hTbf[((long long)b * NP + n) * Cp + c] = u;
}

__global__ void xx_kernel(const float* __restrict__ h, float* __restrict__ xx, int C)
{
  int i = blockIdx.x * 256 + threadIdx.x;
  if (i >= BB * NP) return;
  int b = i >> 10, n = i & 1023;
  const float* hb = h + (long long)b * C * NP;
  float s = 0.f;
  for (int c = 0; c < C; ++c) { float t = hb[c * NP + n]; s += t * t; }
  xx[i] = s;
}

__global__ __launch_bounds__(32) void topk_kernel(
    const float* __restrict__ inner, const float* __restrict__ xx, int* __restrict__ idx)
{
  int gb = blockIdx.x;
  int b = gb >> 10, n = gb & 1023;
  int lane = threadIdx.x;
  const float* row = inner + ((long long)b * NP + n) * NP;
  const float* xb = xx + b * NP;
  float xi = xb[n];
  float v[32];
#pragma unroll 4
  for (int t = 0; t < 32; ++t) {
    int c = t * 32 + lane;
    v[t] = 2.f * row[c] - xi - xb[c];
  }
  int* orow = idx + ((long long)b * NP + n) * KNN;
  for (int it = 0; it < KNN; ++it) {
    float bestv = -3.0e38f;
    int besti = 0;
#pragma unroll
    for (int t = 0; t < 32; ++t) {
      int c = t * 32 + lane;
      if (v[t] > bestv || (v[t] == bestv && c < besti)) { bestv = v[t]; besti = c; }
    }
#pragma unroll
    for (int off = 16; off > 0; off >>= 1) {
      float ov = __shfl_xor(bestv, off);
      int oi = __shfl_xor(besti, off);
      if (ov > bestv || (ov == bestv && oi < besti)) { bestv = ov; besti = oi; }
    }
    if (lane == (besti & 31)) v[besti >> 5] = -3.0e38f;
    if (lane == 0) orow[it] = besti;
  }
}

__global__ void prepw_kernel(const float* __restrict__ w, unsigned short* __restrict__ wbf,
                             int O, int C2)
{
  int i = blockIdx.x * 256 + threadIdx.x;
  if (i >= 128 * 256) return;
  int m = i >> 8, k = i & 255;
  float v = (m < O && k < C2) ? w[m * C2 + k] : 0.f;
  wbf[i] = f2bf(v);
}

__global__ void finstats_kernel(const float* gsum, const float* gsumsq,
                                float* mean, float* inv, int O)
{
  int o = threadIdx.x;
  if (o >= O) return;
  const float cnt = (float)(BB * NP * KNN);
  float m = gsum[o] / cnt;
  float var = gsumsq[o] / cnt - m * m;
  mean[o] = m;
  inv[o] = rsqrtf(var + 1e-5f);
}

__global__ void fillf_kernel(float* p, float v, long long nElem)
{
  long long i = (long long)blockIdx.x * 256 + threadIdx.x;
  if (i < nElem) p[i] = v;
}

__global__ void ytoxt_kernel(const float* __restrict__ yout,
                             unsigned short* __restrict__ xtbf, int O, int Ep)
{
  long long i = (long long)blockIdx.x * 256 + threadIdx.x;
  long long tot = (long long)BB * NP * Ep;
  if (i >= tot) return;
  int e = (int)(i % Ep);
  long long r = i / Ep;
  int n = (int)(r % NP);
  int b = (int)(r / NP);
  float v = (e < O) ? yout[((long long)b * O + e) * NP + n] : 0.f;
  xtbf[i] = f2bf(v);
}

__global__ void prepwin_kernel(const float* __restrict__ win,
                               unsigned short* __restrict__ wbf, int E, int Ep, int N3)
{
  int i = blockIdx.x * 256 + threadIdx.x;
  if (i >= Ep * N3) return;
  int k = i / N3, j = i % N3;
  float v = (k < E && j < 3 * E) ? win[j * E + k] : 0.f;
  wbf[i] = f2bf(v);
}

__global__ __launch_bounds__(32) void attn_kernel(
    const float* __restrict__ qkv, const float* __restrict__ b_in,
    unsigned short* __restrict__ attbf, int E, int H, int N3, int Ep)
{
  int n = blockIdx.x, hh = blockIdx.y;
  int l = threadIdx.x;
  int d = E / H;
  if (l >= BB) return;
  float scale = rsqrtf((float)d);
  float qv[32];
  for (int t = 0; t < d; ++t)
    qv[t] = (qkv[((long long)l * NP + n) * N3 + hh * d + t] + b_in[hh * d + t]) * scale;
  float sc[BB];
  for (int m = 0; m < BB; ++m) {
    float s = 0.f;
    for (int t = 0; t < d; ++t)
      s += qv[t] * (qkv[((long long)m * NP + n) * N3 + E + hh * d + t] + b_in[E + hh * d + t]);
    sc[m] = s;
  }
  float mx = sc[0];
  for (int m = 1; m < BB; ++m) mx = fmaxf(mx, sc[m]);
  float den = 0.f;
  for (int m = 0; m < BB; ++m) { sc[m] = __expf(sc[m] - mx); den += sc[m]; }
  float rd = 1.f / den;
  for (int t = 0; t < d; ++t) {
    float o = 0.f;
    for (int m = 0; m < BB; ++m)
      o += sc[m] * (qkv[((long long)m * NP + n) * N3 + 2 * E + hh * d + t] + b_in[2 * E + hh * d + t]);
    attbf[((long long)l * NP + n) * Ep + hh * d + t] = f2bf(o * rd);
  }
}

__global__ void prepwout_kernel(const float* __restrict__ wout,
                                unsigned short* __restrict__ wbf, int E, int Ep, int No)
{
  int i = blockIdx.x * 256 + threadIdx.x;
  if (i >= Ep * No) return;
  int k = i / No, j = i % No;
  float v = (k < E && j < E) ? wout[j * E + k] : 0.f;
  wbf[i] = f2bf(v);
}

__global__ void resid_kernel(const float* __restrict__ proj, const float* __restrict__ bout,
                             const float* __restrict__ yout, float* __restrict__ hnext,
                             float* __restrict__ feats, int O, int No, int choff)
{
  long long i = (long long)blockIdx.x * 256 + threadIdx.x;
  long long tot = (long long)BB * O * NP;
  if (i >= tot) return;
  int n = (int)(i % NP);
  long long r = i / NP;
  int o = (int)(r % O);
  int b = (int)(r / O);
  float v = proj[((long long)b * NP + n) * No + o] + bout[o] + yout[i];
  hnext[i] = v;
  feats[((long long)b * FEATS_CH + choff + o) * NP + n] = v;
}

__global__ void featsbf_kernel(const float* __restrict__ feats, unsigned short* __restrict__ fbf)
{
  long long i = (long long)blockIdx.x * 256 + threadIdx.x;
  long long tot = (long long)BB * 768 * NP;
  if (i >= tot) return;
  int n = (int)(i % NP);
  long long r = i / NP;
  int c = (int)(r % 768);
  int b = (int)(r / 768);
  float v = (c < FEATS_CH) ? feats[((long long)b * FEATS_CH + c) * NP + n] : 0.f;
  fbf[i] = f2bf(v);
}

__global__ void prepc5_kernel(const float* __restrict__ w, unsigned short* __restrict__ wbf)
{
  long long i = (long long)blockIdx.x * 256 + threadIdx.x;
  if (i >= 1024LL * 768) return;
  int k = (int)(i % 768);
  int m = (int)(i / 768);
  wbf[i] = f2bf(k < FEATS_CH ? w[(long long)m * FEATS_CH + k] : 0.f);
}

__global__ __launch_bounds__(256) void bn2dstats_kernel(const float* __restrict__ z,
                                                        float* mean, float* inv)
{
  int o = blockIdx.x;
  __shared__ float ss[256], sq[256];
  float s = 0.f, q = 0.f;
  for (int i = threadIdx.x; i < BB * NP; i += 256) {
    int b = i >> 10, n = i & 1023;
    float v = z[((long long)b * EMB + o) * NP + n];
    s += v; q += v * v;
  }
  ss[threadIdx.x] = s; sq[threadIdx.x] = q;
  __syncthreads();
  for (int st = 128; st > 0; st >>= 1) {
    if (threadIdx.x < st) { ss[threadIdx.x] += ss[threadIdx.x + st]; sq[threadIdx.x] += sq[threadIdx.x + st]; }
    __syncthreads();
  }
  if (threadIdx.x == 0) {
    float m = ss[0] / (float)(BB * NP);
    float var = sq[0] / (float)(BB * NP) - m * m;
    mean[o] = m; inv[o] = rsqrtf(var + 1e-5f);
  }
}

__global__ __launch_bounds__(256) void pool_kernel(const float* __restrict__ z,
                                                   const float* mean, const float* inv,
                                                   const float* g, const float* beta,
                                                   float* __restrict__ pooled)
{
  int o = blockIdx.x, b = blockIdx.y;
  __shared__ float sm[256], sa[256];
  float mx = -3.0e38f, s = 0.f;
  float m = mean[o], iv = inv[o], gg = g[o], bb2 = beta[o];
  for (int n = threadIdx.x; n < NP; n += 256) {
    float v = leakyf(gg * (z[((long long)b * EMB + o) * NP + n] - m) * iv + bb2);
    mx = fmaxf(mx, v); s += v;
  }
  sm[threadIdx.x] = mx; sa[threadIdx.x] = s;
  __syncthreads();
  for (int st = 128; st > 0; st >>= 1) {
    if (threadIdx.x < st) { sm[threadIdx.x] = fmaxf(sm[threadIdx.x], sm[threadIdx.x + st]); sa[threadIdx.x] += sa[threadIdx.x + st]; }
    __syncthreads();
  }
  if (threadIdx.x == 0) {
    pooled[b * 2048 + o] = sm[0];
    pooled[b * 2048 + EMB + o] = sa[0] / (float)NP;
  }
}

__global__ __launch_bounds__(256) void fc_kernel(const float* __restrict__ in,
                                                 const float* __restrict__ W,
                                                 const float* __restrict__ bias,
                                                 float* __restrict__ out, int Kd, int F)
{
  int j = blockIdx.x, b = blockIdx.y;
  __shared__ float ss[256];
  float s = 0.f;
  for (int k = threadIdx.x; k < Kd; k += 256) s += in[b * Kd + k] * W[(long long)j * Kd + k];
  ss[threadIdx.x] = s;
  __syncthreads();
  for (int st = 128; st > 0; st >>= 1) {
    if (threadIdx.x < st) ss[threadIdx.x] += ss[threadIdx.x + st];
    __syncthreads();
  }
  if (threadIdx.x == 0) out[b * F + j] = ss[0] + (bias ? bias[j] : 0.f);
}

__global__ void bn1d_kernel(float* __restrict__ x, const float* g, const float* beta, int F)
{
  int j = blockIdx.x * blockDim.x + threadIdx.x;
  if (j >= F) return;
  float s = 0.f, q = 0.f;
  for (int b = 0; b < BB; ++b) { float v = x[b * F + j]; s += v; q += v * v; }
  float m = s / BB, var = q / BB - m * m;
  float iv = rsqrtf(var + 1e-5f);
  for (int b = 0; b < BB; ++b) x[b * F + j] = leakyf(g[j] * (x[b * F + j] - m) * iv + beta[j]);
}

// ---------------------------------------------------------------------------
extern "C" void kernel_launch(void* const* d_in, const int* in_sizes, int n_in,
                              void* d_out, int out_size, void* d_ws, size_t ws_size,
                              hipStream_t stream)
{
  (void)in_sizes; (void)n_in; (void)out_size; (void)ws_size;
  static const int CIN_[7]   = {3, 4, 8, 16, 32, 64, 128};
  static const int COUT_[7]  = {4, 8, 16, 32, 64, 128, 128};
  static const int HEADS_[7] = {1, 4, 4, 4, 4, 4, 4};
  static const int CHOFF_[7] = {0, 4, 12, 28, 60, 124, 252};

  const float* x = (const float*)d_in[0];
  const float* conv5_w = (const float*)d_in[99];
  const float* conv5_g = (const float*)d_in[100];
  const float* conv5_b = (const float*)d_in[101];
  const float* l1_w = (const float*)d_in[102];
  const float* bn6_g = (const float*)d_in[103];
  const float* bn6_b = (const float*)d_in[104];
  const float* l2_w = (const float*)d_in[105];
  const float* l2_b = (const float*)d_in[106];
  const float* bn7_g = (const float*)d_in[107];
  const float* bn7_b = (const float*)d_in[108];
  const float* l3_w = (const float*)d_in[109];
  const float* l3_b = (const float*)d_in[110];

  char* wsp = (char*)d_ws;
  size_t off = 0;
  auto alloc = [&](size_t bytes) -> void* {
    void* p = wsp + off;
    off = (off + bytes + 255) & ~(size_t)255;
    return p;
  };

  float* inner = (float*)alloc((size_t)BB * NP * NP * 4);        // reused as z
  unsigned short* hbf  = (unsigned short*)alloc((size_t)BB * 128 * NP * 2);
  unsigned short* hTbf = (unsigned short*)alloc((size_t)BB * NP * 128 * 2);
  float* xx = (float*)alloc((size_t)BB * NP * 4);
  int* idx = (int*)alloc((size_t)BB * NP * KNN * 4);
  unsigned short* wbf = (unsigned short*)alloc(128 * 256 * 2);
  float* gsum = (float*)alloc(128 * 4);
  float* gsumsq = (float*)alloc(128 * 4);
  float* bmean = (float*)alloc(128 * 4);
  float* binv = (float*)alloc(128 * 4);
  float* yout = (float*)alloc((size_t)BB * 128 * NP * 4);
  unsigned short* xtbf = (unsigned short*)alloc((size_t)BB * NP * 128 * 2);
  unsigned short* winbf = (unsigned short*)alloc(128 * 384 * 2);
  float* qkv = (float*)alloc((size_t)BB * NP * 384 * 4);
  unsigned short* attbf = (unsigned short*)alloc((size_t)BB * NP * 128 * 2);
  unsigned short* woutbf = (unsigned short*)alloc(128 * 128 * 2);
  float* proj = (float*)alloc((size_t)BB * NP * 128 * 4);
  float* hA = (float*)alloc((size_t)BB * 128 * NP * 4);
  float* hB = (float*)alloc((size_t)BB * 128 * NP * 4);
  float* feats = (float*)alloc((size_t)BB * FEATS_CH * NP * 4);
  unsigned short* fbf = (unsigned short*)alloc((size_t)BB * 768 * NP * 2);
  unsigned short* c5wbf = (unsigned short*)alloc((size_t)1024 * 768 * 2);
  float* mean2 = (float*)alloc(1024 * 4);
  float* inv2 = (float*)alloc(1024 * 4);
  float* pooled = (float*)alloc(BB * 2048 * 4);
  float* fc1 = (float*)alloc(BB * 512 * 4);
  float* fc2 = (float*)alloc(BB * 256 * 4);
  float* zbuf = inner;

  for (int j = 0; j < 14; ++j) {
    int bi = j % 7;
    int C = CIN_[bi], O = COUT_[bi], H = HEADS_[bi];
    int choff = (j < 7 ? 0 : 380) + CHOFF_[bi];
    int base = 1 + j * 7;
    const float* conv_w = (const float*)d_in[base + 0];
    const float* bn_g   = (const float*)d_in[base + 1];
    const float* bn_b   = (const float*)d_in[base + 2];
    const float* w_in   = (const float*)d_in[base + 3];
    const float* b_in   = (const float*)d_in[base + 4];
    const float* w_out  = (const float*)d_in[base + 5];
    const float* b_out  = (const float*)d_in[base + 6];

    const float* hcur = (bi == 0) ? x : ((bi & 1) ? hA : hB);
    float* hout = (bi & 1) ? hB : hA;

    int Cp  = (C + 31) & ~31;
    int KdP = (2 * C + 31) & ~31;
    int Ep  = (O + 31) & ~31;
    int N3  = (3 * O + 63) & ~63;
    int No  = (O + 63) & ~63;

    // --- kNN: Gram matrix (WMMA) + top-k selection ---
    long long tot1 = (long long)BB * Cp * NP;
    convgram_kernel<<<(int)((tot1 + 255) / 256), 256, 0, stream>>>(hcur, hbf, hTbf, C, Cp);
    xx_kernel<<<(BB * NP + 255) / 256, 256, 0, stream>>>(hcur, xx, C);
    {
      dim3 g(NP / GT_N, NP / GT_M, BB);
      gemm_bf16_kernel<<<g, 256, 0, stream>>>(hTbf, hbf, inner, NP, NP, Cp,
          (long long)NP * Cp, (long long)Cp * NP, (long long)NP * NP);
    }
    topk_kernel<<<BB * NP, 32, 0, stream>>>(inner, xx, idx);

    // --- edge conv: stats pass + BN/leaky/max-over-k pass ---
    prepw_kernel<<<128, 256, 0, stream>>>(conv_w, wbf, O, 2 * C);
    hipMemsetAsync(gsum, 0, 128 * 4, stream);
    hipMemsetAsync(gsumsq, 0, 128 * 4, stream);
    {
      dim3 g(NP * KNN / 64, BB);
      edgeconv_kernel<<<g, 256, 0, stream>>>(hcur, idx, wbf, C, O, KdP, 1,
          gsum, gsumsq, nullptr, nullptr, nullptr, nullptr, nullptr);
      finstats_kernel<<<1, 128, 0, stream>>>(gsum, gsumsq, bmean, binv, O);
      long long ytot = (long long)BB * O * NP;
      fillf_kernel<<<(int)((ytot + 255) / 256), 256, 0, stream>>>(yout, -3.0e38f, ytot);
      edgeconv_kernel<<<g, 256, 0, stream>>>(hcur, idx, wbf, C, O, KdP, 2,
          gsum, gsumsq, bmean, binv, bn_g, bn_b, yout);
    }

    // --- MHA: qkv GEMM (WMMA), softmax over L=8, out-proj GEMM, residual ---
    long long xtot = (long long)BB * NP * Ep;
    ytoxt_kernel<<<(int)((xtot + 255) / 256), 256, 0, stream>>>(yout, xtbf, O, Ep);
    prepwin_kernel<<<(Ep * N3 + 255) / 256, 256, 0, stream>>>(w_in, winbf, O, Ep, N3);
    {
      dim3 g(N3 / GT_N, (BB * NP) / GT_M, 1);
      gemm_bf16_kernel<<<g, 256, 0, stream>>>(xtbf, winbf, qkv, BB * NP, N3, Ep, 0, 0, 0);
    }
    hipMemsetAsync(attbf, 0, (size_t)BB * NP * Ep * 2, stream);
    {
      dim3 g(NP, H);
      attn_kernel<<<g, 32, 0, stream>>>(qkv, b_in, attbf, O, H, N3, Ep);
    }
    prepwout_kernel<<<(Ep * No + 255) / 256, 256, 0, stream>>>(w_out, woutbf, O, Ep, No);
    {
      dim3 g(No / GT_N, (BB * NP) / GT_M, 1);
      gemm_bf16_kernel<<<g, 256, 0, stream>>>(attbf, woutbf, proj, BB * NP, No, Ep, 0, 0, 0);
    }
    long long rtot = (long long)BB * O * NP;
    resid_kernel<<<(int)((rtot + 255) / 256), 256, 0, stream>>>(proj, b_out, yout, hout,
                                                                feats, O, No, choff);
  }

  // --- conv5 (WMMA) + BN + pool + MLP head ---
  long long ftot = (long long)BB * 768 * NP;
  featsbf_kernel<<<(int)((ftot + 255) / 256), 256, 0, stream>>>(feats, fbf);
  prepc5_kernel<<<(int)((1024LL * 768 + 255) / 256), 256, 0, stream>>>(conv5_w, c5wbf);
  {
    dim3 g(NP / GT_N, EMB / GT_M, BB);
    gemm_bf16_kernel<<<g, 256, 0, stream>>>(c5wbf, fbf, zbuf, EMB, NP, 768,
        0LL, (long long)768 * NP, (long long)EMB * NP);
  }
  bn2dstats_kernel<<<EMB, 256, 0, stream>>>(zbuf, mean2, inv2);
  { dim3 g(EMB, BB); pool_kernel<<<g, 256, 0, stream>>>(zbuf, mean2, inv2, conv5_g, conv5_b, pooled); }
  { dim3 g(512, BB); fc_kernel<<<g, 256, 0, stream>>>(pooled, l1_w, nullptr, fc1, 2048, 512); }
  bn1d_kernel<<<2, 256, 0, stream>>>(fc1, bn6_g, bn6_b, 512);
  { dim3 g(256, BB); fc_kernel<<<g, 256, 0, stream>>>(fc1, l2_w, l2_b, fc2, 512, 256); }
  bn1d_kernel<<<1, 256, 0, stream>>>(fc2, bn7_g, bn7_b, 256);
  { dim3 g(NCLS, BB); fc_kernel<<<g, 256, 0, stream>>>(fc2, l3_w, l3_b, (float*)d_out, 256, NCLS); }
}
